// MultiHeadAttention_13142599926456
// MI455X (gfx1250) — compile-verified, matrix operand order checked by reading, not run on hardware
//
#include <hip/hip_runtime.h>
#include <hip/hip_bf16.h>

#define D_MODEL 768
#define NH 12
#define HD 64
#define TT 2048
#define BB 4
#define MTOT (BB * TT)   // 8192

typedef __attribute__((ext_vector_type(16))) _Float16 v16h;
typedef __attribute__((ext_vector_type(8)))  float    v8f;
typedef __attribute__((ext_vector_type(4)))  float    f32x4;

union FragU { v16h v; f32x4 q[2]; };

// Load a 16x32 f16 WMMA fragment (A or B, K-major rows).
// Per lane: halfs 0..7 at p[0..7]   (K = 8*cg + 0..7),
//           halfs 8..15 at p[16..23] (K = 16 + 8*cg + 0..7).
// Caller passes p = rowbase + 8*cg; addresses are 16B aligned by construction.
__device__ __forceinline__ v16h load_frag(const _Float16* p) {
  FragU u;
  u.q[0] = *(const f32x4*)(p);
  u.q[1] = *(const f32x4*)(p + 16);
  return u.v;
}

__device__ __forceinline__ v8f wmma_f16(v16h a, v16h b, v8f c) {
  return __builtin_amdgcn_wmma_f32_16x16x32_f16(false, a, false, b, (short)0, c,
                                                false, false);
}

__device__ __forceinline__ v8f vzero8() {
  v8f z;
#pragma unroll
  for (int i = 0; i < 8; ++i) z[i] = 0.0f;
  return z;
}

// ---------------------------------------------------------------- convert
__global__ void cvt_f32_f16(const float* __restrict__ src,
                            _Float16* __restrict__ dst, int n) {
  int i = blockIdx.x * blockDim.x + threadIdx.x;
  int stride = gridDim.x * blockDim.x;
  for (; i < n; i += stride) dst[i] = (_Float16)src[i];
}

// ---------------------------------------------------------------- GEMM
// C[m,n] = sum_k A[m,k] * W[n,k]   (A: [MTOT,K] f16 row-major, W: [N,K] f16)
// Wave tile 32M x 64N; K-loop unrolled x2 as a ping-pong (two named fragment
// sets, no cross-set copies). Requires K % 64 == 0 and K >= 64.
// mode 0: out16[((b*NH+h)*TT + t)*HD + d] = val*oscale   (Q/K layout)
// mode 2: out16[((b*NH+h)*HD + d)*TT + t] = val          (V transposed)
// mode 3: out32[m*Nld + n] = val + bias[n]               (final output)
__global__ __launch_bounds__(256) void gemm_f16(
    const _Float16* __restrict__ A, const _Float16* __restrict__ W, int K,
    int mode, float oscale, const float* __restrict__ bias,
    _Float16* __restrict__ out16, float* __restrict__ out32, int Nld) {
  const int tid  = threadIdx.x;
  const int wid  = tid >> 5;
  const int lane = tid & 31;
  const int r    = lane & 15;   // row-of-A / col-of-B / col-of-C
  const int cg   = lane >> 4;   // K-group
  const int wm   = wid >> 1;    // 4 waves in M
  const int wn   = wid & 1;     // 2 waves in N
  const int M0 = blockIdx.x * 128 + wm * 32;
  const int N0 = blockIdx.y * 128 + wn * 64;

  const _Float16* arow0 = A + (size_t)(M0 + r) * K + 8 * cg;
  const _Float16* arow1 = arow0 + (size_t)16 * K;
  const _Float16* wrow  = W + (size_t)(N0 + r) * K + 8 * cg;

  v8f acc[2][4];
#pragma unroll
  for (int mi = 0; mi < 2; ++mi)
#pragma unroll
    for (int nt = 0; nt < 4; ++nt) acc[mi][nt] = vzero8();

  // ping-pong fragment sets
  v16h a0A, a1A, bA[4], a0B, a1B, bB[4];

  // prologue: set A <- k = 0
  a0A = load_frag(arow0);
  a1A = load_frag(arow1);
#pragma unroll
  for (int nt = 0; nt < 4; ++nt) bA[nt] = load_frag(wrow + (size_t)nt * 16 * K);

  int k0 = 0;
  for (; k0 + 64 < K; k0 += 64) {
    // load set B <- k0+32
    a0B = load_frag(arow0 + k0 + 32);
    a1B = load_frag(arow1 + k0 + 32);
#pragma unroll
    for (int nt = 0; nt < 4; ++nt)
      bB[nt] = load_frag(wrow + (size_t)nt * 16 * K + k0 + 32);
    // compute set A (k0)
#pragma unroll
    for (int nt = 0; nt < 4; ++nt) acc[0][nt] = wmma_f16(a0A, bA[nt], acc[0][nt]);
#pragma unroll
    for (int nt = 0; nt < 4; ++nt) acc[1][nt] = wmma_f16(a1A, bA[nt], acc[1][nt]);
    // load set A <- k0+64
    a0A = load_frag(arow0 + k0 + 64);
    a1A = load_frag(arow1 + k0 + 64);
#pragma unroll
    for (int nt = 0; nt < 4; ++nt)
      bA[nt] = load_frag(wrow + (size_t)nt * 16 * K + k0 + 64);
    // compute set B (k0+32)
#pragma unroll
    for (int nt = 0; nt < 4; ++nt) acc[0][nt] = wmma_f16(a0B, bB[nt], acc[0][nt]);
#pragma unroll
    for (int nt = 0; nt < 4; ++nt) acc[1][nt] = wmma_f16(a1B, bB[nt], acc[1][nt]);
  }
  // k0 == K-64: set A holds k0; load set B <- K-32, then compute both
  a0B = load_frag(arow0 + k0 + 32);
  a1B = load_frag(arow1 + k0 + 32);
#pragma unroll
  for (int nt = 0; nt < 4; ++nt)
    bB[nt] = load_frag(wrow + (size_t)nt * 16 * K + k0 + 32);
#pragma unroll
  for (int nt = 0; nt < 4; ++nt) acc[0][nt] = wmma_f16(a0A, bA[nt], acc[0][nt]);
#pragma unroll
  for (int nt = 0; nt < 4; ++nt) acc[1][nt] = wmma_f16(a1A, bA[nt], acc[1][nt]);
#pragma unroll
  for (int nt = 0; nt < 4; ++nt) acc[0][nt] = wmma_f16(a0B, bB[nt], acc[0][nt]);
#pragma unroll
  for (int nt = 0; nt < 4; ++nt) acc[1][nt] = wmma_f16(a1B, bB[nt], acc[1][nt]);

  // -------- epilogue: wave-constant bases, linear offsets ----------------
  // Wave's 32 M-rows lie in one batch (M0 % 32 == 0, TT % 32 == 0);
  // its 64 N-cols lie in one head (N0 % 64 == 0, HD == 64).
  if (mode == 3) {
    float bv[4];
#pragma unroll
    for (int nt = 0; nt < 4; ++nt) bv[nt] = bias[N0 + nt * 16 + r];
    float* op = out32 + (size_t)(M0 + 8 * cg) * Nld + N0 + r;
#pragma unroll
    for (int mi = 0; mi < 2; ++mi)
#pragma unroll
      for (int nt = 0; nt < 4; ++nt)
#pragma unroll
        for (int j = 0; j < 8; ++j)
          op[(size_t)(mi * 16 + j) * Nld + nt * 16] =
              acc[mi][nt][j] * oscale + bv[nt];
  } else {
    const int bb = M0 / TT;
    const int t0 = M0 % TT;
    const int h  = N0 / HD;
    if (mode == 2) {   // V^T: [B,H,HD,T]
      _Float16* op =
          out16 + ((((size_t)bb * NH + h) * HD + r) * TT) + t0 + 8 * cg;
#pragma unroll
      for (int mi = 0; mi < 2; ++mi)
#pragma unroll
        for (int nt = 0; nt < 4; ++nt)
#pragma unroll
          for (int j = 0; j < 8; ++j)
            op[(size_t)(nt * 16) * TT + mi * 16 + j] =
                (_Float16)(acc[mi][nt][j] * oscale);
    } else {           // Q/K: [B,H,T,HD]
      _Float16* op =
          out16 + (((size_t)bb * NH + h) * TT + t0 + 8 * cg) * HD + r;
#pragma unroll
      for (int mi = 0; mi < 2; ++mi)
#pragma unroll
        for (int nt = 0; nt < 4; ++nt)
#pragma unroll
          for (int j = 0; j < 8; ++j)
            op[(size_t)(mi * 16 + j) * HD + nt * 16] =
                (_Float16)(acc[mi][nt][j] * oscale);
    }
  }
}

// ---------------------------------------------------------------- flash attn
// Q,K: [B,H,T,HD] f16 (Q pre-scaled by 1/sqrt(HD)); Vt: [B,H,HD,T] f16.
// Each wave owns one 16-query tile; online softmax over 32-key blocks.
// V fragments load at block top so they fly during the softmax section.
__global__ __launch_bounds__(128) void flash_attn(
    const _Float16* __restrict__ Q, const _Float16* __restrict__ K,
    const _Float16* __restrict__ Vt, _Float16* __restrict__ O) {
  __shared__ _Float16 pbuf[4][16][32];   // per-wave P transpose staging

  const int tid  = threadIdx.x;
  const int wid  = tid >> 5;
  const int lane = tid & 31;
  const int r    = lane & 15;
  const int cg   = lane >> 4;

  const int gw = blockIdx.x * 4 + wid;   // global wave = (bh, qtile)
  const int bh = gw >> 7;                // TT/16 = 128 q-tiles per head
  const int qt = gw & 127;
  const int q0 = qt * 16;

  const _Float16* Qh = Q  + (size_t)bh * TT * HD;
  const _Float16* Kh = K  + (size_t)bh * TT * HD;
  const _Float16* Vh = Vt + (size_t)bh * HD * TT;

  // Q A-fragments for d=0..31 and d=32..63 (loaded once)
  const _Float16* qbase = Qh + (size_t)(q0 + r) * HD + 8 * cg;
  v16h qa0 = load_frag(qbase);
  v16h qa1 = load_frag(qbase + 32);

  float mrow[8], lrow[8];
  v8f o[4];
#pragma unroll
  for (int nt = 0; nt < 4; ++nt) o[nt] = vzero8();
#pragma unroll
  for (int j = 0; j < 8; ++j) { mrow[j] = -__builtin_inff(); lrow[j] = 0.0f; }

  const int nblk = q0 / 32 + 1;   // key blocks of 32 covering keys <= q0+15
  for (int kb = 0; kb < nblk; ++kb) {
    const int k0 = kb * 32;

    // issue ALL loads for this block first: K frags + V frags
    const _Float16* kp = Kh + (size_t)(k0 + r) * HD + 8 * cg;
    v16h b0 = load_frag(kp);
    v16h b1 = load_frag(kp + 32);
    v16h b2 = load_frag(kp + 16 * HD);
    v16h b3 = load_frag(kp + 16 * HD + 32);
    v16h vb[4];
#pragma unroll
    for (int nt = 0; nt < 4; ++nt)
      vb[nt] = load_frag(Vh + (size_t)(nt * 16 + r) * TT + k0 + 8 * cg);

    // S = Q · K^T for 32 keys -> two 16x16 f32 tiles
    v8f s0 = wmma_f16(qa0, b0, vzero8());
    s0 = wmma_f16(qa1, b1, s0);
    v8f s1 = wmma_f16(qa0, b2, vzero8());
    s1 = wmma_f16(qa1, b3, s1);

    // causal mask on diagonal-crossing blocks (selects, no divergence)
    if (k0 + 31 > q0) {
#pragma unroll
      for (int j = 0; j < 8; ++j) {
        const int rowq = q0 + j + 8 * cg;
        s0[j] = (k0 + r      > rowq) ? -1.0e30f : s0[j];
        s1[j] = (k0 + 16 + r > rowq) ? -1.0e30f : s1[j];
      }
    }

    // row max across the 32 keys (16 cols live across a 16-lane group)
    float bm[8];
#pragma unroll
    for (int j = 0; j < 8; ++j) bm[j] = fmaxf(s0[j], s1[j]);
#pragma unroll
    for (int mk = 1; mk < 16; mk <<= 1) {
#pragma unroll
      for (int j = 0; j < 8; ++j)
        bm[j] = fmaxf(bm[j], __shfl_xor(bm[j], mk, 16));
    }

    float ps[8];
#pragma unroll
    for (int j = 0; j < 8; ++j) {
      const float mn = fmaxf(mrow[j], bm[j]);
      const float rf = __expf(mrow[j] - mn);
      mrow[j] = mn;
      const float e0 = __expf(s0[j] - mn);
      const float e1 = __expf(s1[j] - mn);
      pbuf[wid][j + 8 * cg][r]      = (_Float16)e0;   // C-layout -> LDS
      pbuf[wid][j + 8 * cg][r + 16] = (_Float16)e1;
      ps[j] = e0 + e1;
      lrow[j] *= rf;
      o[0][j] *= rf; o[1][j] *= rf; o[2][j] *= rf; o[3][j] *= rf;
    }
#pragma unroll
    for (int mk = 1; mk < 16; mk <<= 1) {
#pragma unroll
      for (int j = 0; j < 8; ++j) ps[j] += __shfl_xor(ps[j], mk, 16);
    }
#pragma unroll
    for (int j = 0; j < 8; ++j) lrow[j] += ps[j];

    // P as A-fragment (16 queries x 32 keys) from LDS; same-wave DS in order
    v16h pa = load_frag(&pbuf[wid][r][8 * cg]);

    // O += P · V  (V frags already resident)
#pragma unroll
    for (int nt = 0; nt < 4; ++nt) o[nt] = wmma_f16(pa, vb[nt], o[nt]);
  }

  float inv[8];
#pragma unroll
  for (int j = 0; j < 8; ++j) inv[j] = 1.0f / lrow[j];

  // output base: row m = b*TT + q0 + j + 8cg, col = h*64 + nt*16 + r
  const int b = bh / NH, h = bh % NH;
  _Float16* op =
      O + (size_t)(b * TT + q0 + 8 * cg) * D_MODEL + h * HD + r;
#pragma unroll
  for (int nt = 0; nt < 4; ++nt)
#pragma unroll
    for (int j = 0; j < 8; ++j)
      op[(size_t)j * D_MODEL + nt * 16] = (_Float16)(o[nt][j] * inv[j]);
}

// ---------------------------------------------------------------- launch
extern "C" void kernel_launch(void* const* d_in, const int* in_sizes, int n_in,
                              void* d_out, int out_size, void* d_ws,
                              size_t ws_size, hipStream_t stream) {
  (void)in_sizes; (void)n_in; (void)out_size; (void)ws_size;
  const float* x  = (const float*)d_in[0];
  const float* Wq = (const float*)d_in[1];
  const float* Wk = (const float*)d_in[2];
  const float* Wv = (const float*)d_in[3];
  const float* Wp = (const float*)d_in[4];
  const float* bp = (const float*)d_in[5];
  float* out = (float*)d_out;

  char* ws = (char*)d_ws;
  const size_t SX = (size_t)MTOT * D_MODEL * sizeof(_Float16);  // 12.6 MB
  const size_t SW = (size_t)D_MODEL * D_MODEL * sizeof(_Float16);
  _Float16* xh  = (_Float16*)(ws);
  _Float16* wqh = (_Float16*)(ws + SX);
  _Float16* wkh = (_Float16*)(ws + SX + SW);
  _Float16* wvh = (_Float16*)(ws + SX + 2 * SW);
  _Float16* wph = (_Float16*)(ws + SX + 3 * SW);
  _Float16* qb  = (_Float16*)(ws + SX + 4 * SW);           // [B,H,T,HD]
  _Float16* kbf = (_Float16*)(ws + 2 * SX + 4 * SW);       // [B,H,T,HD]
  _Float16* vtb = (_Float16*)(ws + 3 * SX + 4 * SW);       // [B,H,HD,T]
  _Float16* ao  = (_Float16*)(ws + 4 * SX + 4 * SW);       // [MTOT,768]

  cvt_f32_f16<<<1024, 256, 0, stream>>>(x, xh, MTOT * D_MODEL);
  cvt_f32_f16<<<256, 256, 0, stream>>>(Wq, wqh, D_MODEL * D_MODEL);
  cvt_f32_f16<<<256, 256, 0, stream>>>(Wk, wkh, D_MODEL * D_MODEL);
  cvt_f32_f16<<<256, 256, 0, stream>>>(Wv, wvh, D_MODEL * D_MODEL);
  cvt_f32_f16<<<256, 256, 0, stream>>>(Wp, wph, D_MODEL * D_MODEL);

  dim3 gg(MTOT / 128, D_MODEL / 128);
  const float scale = 0.125f;   // 1/sqrt(64), folded into Q
  gemm_f16<<<gg, 256, 0, stream>>>(xh, wqh, D_MODEL, 0, scale, nullptr, qb,
                                   nullptr, D_MODEL);
  gemm_f16<<<gg, 256, 0, stream>>>(xh, wkh, D_MODEL, 0, 1.0f, nullptr, kbf,
                                   nullptr, D_MODEL);
  gemm_f16<<<gg, 256, 0, stream>>>(xh, wvh, D_MODEL, 2, 1.0f, nullptr, vtb,
                                   nullptr, D_MODEL);

  // BB*NH*(TT/16) waves, 4 per block
  flash_attn<<<(BB * NH * (TT / 16)) / 4, 128, 0, stream>>>(qb, kbf, vtb, ao);

  gemm_f16<<<gg, 256, 0, stream>>>(ao, wph, D_MODEL, 3, 1.0f, bp, nullptr, out,
                                   D_MODEL);
}
